// SpiderSolver_Attention_1D_41369124995293
// MI455X (gfx1250) — matrix operand order repaired and checked by reading, not compiled, not run on hardware
//
#include <hip/hip_runtime.h>

#define HEADS 8
#define DH 64
#define DIM 512
#define ONION 15
#define SC 4
#define N_VELO 28504
#define N_SURF 3682
#define NTOT (N_VELO + N_SURF)
#define SURF_KEEP 3586
#define FIRST_LAP 500
#define NT ((ONION + 1) * SC)     /* 64 */
#define S1 (FIRST_LAP + N_SURF)   /* 4182 */
#define NEG_BIG (-1.0e30f)

typedef __attribute__((ext_vector_type(16))) __bf16 v16bf;
typedef __attribute__((ext_vector_type(8)))  __bf16 v8bf;
typedef __attribute__((ext_vector_type(8)))  float  v8f;

// one-time stores: round-to-nearest-even
static __device__ __forceinline__ unsigned short f2bf_rne(float f) {
  unsigned int u = __builtin_bit_cast(unsigned int, f);
  u += 0x7fffu + ((u >> 16) & 1u);
  return (unsigned short)(u >> 16);
}
// hot staging paths: truncating convert, 1 VALU
static __device__ __forceinline__ unsigned short f2bf(float f) {
  return (unsigned short)(__builtin_bit_cast(unsigned int, f) >> 16);
}
// pack two fp32 -> two bf16 in one v_perm_b32 (bytes 2,3 of each source)
static __device__ __forceinline__ unsigned int pack2(float a, float b) {
  return __builtin_amdgcn_perm(__builtin_bit_cast(unsigned int, b),
                               __builtin_bit_cast(unsigned int, a),
                               0x07060302u);
}
static __device__ __forceinline__ v16bf cat8(v8bf a, v8bf b) {
  return __builtin_shufflevector(a, b, 0, 1, 2, 3, 4, 5, 6, 7,
                                 8, 9, 10, 11, 12, 13, 14, 15);
}
static __device__ __forceinline__ v8f wmma_bf16(v16bf a, v16bf b, v8f c) {
  // D = A(16x32) x B(32x16) + C, f32 accumulate
  return __builtin_amdgcn_wmma_f32_16x16x32_bf16(false, a, false, b, (short)0, c,
                                                 false, false);
}
// ISA fragment maps (CDNA5 ISA §7.12.2):
//  A (16-bit 16x32): lane row M=L&15; elems 0..7 -> K=8*hi..+7, 8..15 -> K=16+8*hi..+7
//  B (16-bit 32x16): lane col N=L&15; elems -> same K map as A (flash_attn uses K=16*hi+s in A and B)
//  C/D (f32 16x16):  lane L, vgpr r -> M=r+8*hi, N=L&15

// ---------------------------------------------------------------------------
// Generic GEMM: C[M,N] = gather(A)[M,K] @ B[K,N] (+bias); fp32 in, fp32/bf16
// out. Register-pipelined double-buffered LDS: global loads of tile k+1 are
// in flight while tile k's WMMAs execute; one barrier per K-step.
// ---------------------------------------------------------------------------
#define BM 128
#define BN 64
#define BK 32
__global__ __launch_bounds__(256)
void gemm_wmma(const float* __restrict__ A, const int* __restrict__ rowidx, int lda,
               const float* __restrict__ B, int ldb, const float* __restrict__ bias,
               void* __restrict__ Cv, int ldc, int M, int N, int K, int obf)
{
  __shared__ alignas(16) unsigned short As[2][BM][BK + 8];  // [m][k], pitch 80B
  __shared__ alignas(16) unsigned short Bs[2][BN][BK + 8];  // transposed [n][k]

  const int tid    = threadIdx.x;
  const int wave   = tid >> 5;
  const int lane   = tid & 31;
  const int lane16 = lane & 15;
  const int hi     = lane >> 4;
  const int m0     = blockIdx.y * BM;
  const int n0     = blockIdx.x * BN;
  const bool fullM = (m0 + BM <= M);
  const bool fullN = (n0 + BN <= N);

  float2 aReg[8];
  float2 bReg[4];

  // ---- global loads of one K-tile into registers (no LDS yet) --------------
  auto loadTiles = [&](int k0) {
    bool fullK = (k0 + BK <= K);
    if (fullM && fullK) {                       // hot path: no exec masking
#pragma unroll
      for (int i = 0; i < 8; ++i) {
        int idx = i * 256 + tid;
        int gm = m0 + (idx >> 4), gk = k0 + 2 * (idx & 15);
        int ar = rowidx ? rowidx[gm] : gm;
        aReg[i] = *(const float2*)(A + (size_t)ar * lda + gk);
      }
    } else {
#pragma unroll
      for (int i = 0; i < 8; ++i) {
        int idx = i * 256 + tid;
        int gm = m0 + (idx >> 4), gk = k0 + 2 * (idx & 15);
        float2 v = {0.0f, 0.0f};
        if (gm < M && gk < K) {                 // K always even
          int ar = rowidx ? rowidx[gm] : gm;
          v = *(const float2*)(A + (size_t)ar * lda + gk);
        }
        aReg[i] = v;
      }
    }
    if (fullN && fullK) {
#pragma unroll
      for (int i = 0; i < 4; ++i) {
        int idx = i * 256 + tid;
        int gk = k0 + 2 * (idx >> 6), gn = n0 + (idx & 63);
        bReg[i].x = B[(size_t)gk * ldb + gn];
        bReg[i].y = B[(size_t)(gk + 1) * ldb + gn];
      }
    } else {
#pragma unroll
      for (int i = 0; i < 4; ++i) {
        int idx = i * 256 + tid;
        int gk = k0 + 2 * (idx >> 6), gn = n0 + (idx & 63);
        float v0 = 0.0f, v1 = 0.0f;
        if (gk < K && gn < N) {
          v0 = B[(size_t)gk * ldb + gn];
          v1 = B[(size_t)(gk + 1) * ldb + gn];
        }
        bReg[i] = {v0, v1};
      }
    }
    // L2 prefetch one further tile ahead (uniform addresses, no masking)
    int kp = k0 + BK;
    if (kp < K && tid < BK)
      __builtin_prefetch(B + (size_t)(kp + tid) * ldb + n0, 0, 1);
  };
  // ---- convert + store registers into LDS buffer (1 v_perm per dword) ------
  auto storeTiles = [&](int buf) {
#pragma unroll
    for (int i = 0; i < 8; ++i) {
      int idx = i * 256 + tid;
      *(unsigned int*)&As[buf][idx >> 4][2 * (idx & 15)] = pack2(aReg[i].x, aReg[i].y);
    }
#pragma unroll
    for (int i = 0; i < 4; ++i) {
      int idx = i * 256 + tid;
      *(unsigned int*)&Bs[buf][idx & 63][2 * (idx >> 6)] = pack2(bReg[i].x, bReg[i].y);
    }
  };

  v8f acc[4] = {};
  const int nk = (K + BK - 1) / BK;

  loadTiles(0);
  storeTiles(0);
  __syncthreads();

  for (int kt = 0; kt < nk; ++kt) {
    int buf = kt & 1;
    if (kt + 1 < nk) loadTiles((kt + 1) * BK);   // loads fly during WMMAs

    const unsigned short* ap = &As[buf][wave * 16 + lane16][0];
    v16bf a = cat8(*(const v8bf*)(ap + 8 * hi),
                   *(const v8bf*)(ap + 16 + 8 * hi));
    v16bf bf4[4];
#pragma unroll
    for (int t = 0; t < 4; ++t) {
      const unsigned short* bp = &Bs[buf][t * 16 + lane16][8 * hi];
      bf4[t] = cat8(*(const v8bf*)bp, *(const v8bf*)(bp + 16));
    }
#pragma unroll
    for (int t = 0; t < 4; ++t) acc[t] = wmma_bf16(a, bf4[t], acc[t]);

    if (kt + 1 < nk) storeTiles(buf ^ 1);
    __syncthreads();
  }

#pragma unroll
  for (int t = 0; t < 4; ++t) {
    int gn = n0 + t * 16 + lane16;
    if (gn >= N) continue;
    float bv = bias ? bias[gn] : 0.0f;
    if (obf) {
      unsigned short* Cb = (unsigned short*)Cv;
#pragma unroll
      for (int r = 0; r < 8; ++r) {
        int gm = m0 + wave * 16 + r + 8 * hi;
        if (gm < M) Cb[(size_t)gm * ldc + gn] = f2bf_rne(acc[t][r] + bv);
      }
    } else {
      float* Cf = (float*)Cv;
#pragma unroll
      for (int r = 0; r < 8; ++r) {
        int gm = m0 + wave * 16 + r + 8 * hi;
        if (gm < M) Cf[(size_t)gm * ldc + gn] = acc[t][r] + bv;
      }
    }
  }
}

// ---------------------------------------------------------------------------
// Flash attention: Q,K,V are bf16 (S, DIM); O fp32; head slice cols [h*64,+64)
// 4 waves x 16 query rows; double-buffered 64-key blocks (K row-major, V
// transposed); next block's global loads overlap this block's WMMAs/softmax.
// ---------------------------------------------------------------------------
__global__ __launch_bounds__(128)
void flash_attn(const unsigned short* __restrict__ Q, const unsigned short* __restrict__ K,
                const unsigned short* __restrict__ V, float* __restrict__ O,
                int S, float scale)
{
  __shared__ alignas(16) unsigned short Ks[2][64][DH + 8];   // [key][d], pitch 144B
  __shared__ alignas(16) unsigned short Vs[2][64][DH + 8];   // [d][key] transposed
  __shared__ alignas(16) unsigned short Ps[4][16][64 + 8];   // per-wave [row][key]

  const int tid    = threadIdx.x;
  const int wave   = tid >> 5;
  const int lane   = tid & 31;
  const int lane16 = lane & 15;
  const int hi     = lane >> 4;
  const int hcol   = blockIdx.y * DH;
  const int qbase  = blockIdx.x * 64 + wave * 16;

  // Q fragments: each run of 8 bf16 = one 16B global load, no conversion
  v16bf Qa[2];
  {
    int qr = qbase + lane16;
    if (qr >= S) qr = 0;
    const unsigned short* qp = Q + (size_t)qr * DIM + hcol;
#pragma unroll
    for (int f = 0; f < 2; ++f)
      Qa[f] = cat8(*(const v8bf*)(qp + 16 * hi + 32 * f),
                   *(const v8bf*)(qp + 8 + 16 * hi + 32 * f));
  }

  uint4 kReg[4];
  unsigned int vReg[16];
  auto loadKV = [&](int kb) {
#pragma unroll
    for (int i = 0; i < 4; ++i) {              // K block: b128 copies
      int idx = i * 128 + tid;
      int r = idx >> 3, q = idx & 7;
      int kr = kb * 64 + r;
      if (kr >= S) kr = 0;
      kReg[i] = *((const uint4*)(K + (size_t)kr * DIM + hcol) + q);
    }
#pragma unroll
    for (int i = 0; i < 16; ++i) {             // V block: b32 reads
      int idx = i * 128 + tid;
      int key = idx >> 5, du = idx & 31;
      int kr = kb * 64 + key;
      if (kr >= S) kr = 0;
      vReg[i] = *(const unsigned int*)(V + (size_t)kr * DIM + hcol + 2 * du);
    }
  };
  auto storeKV = [&](int buf) {
#pragma unroll
    for (int i = 0; i < 4; ++i) {
      int idx = i * 128 + tid;
      *(uint4*)&Ks[buf][idx >> 3][(idx & 7) * 8] = kReg[i];
    }
#pragma unroll
    for (int i = 0; i < 16; ++i) {
      int idx = i * 128 + tid;
      int key = idx >> 5, du = idx & 31;
      Vs[buf][2 * du][key]     = (unsigned short)vReg[i];
      Vs[buf][2 * du + 1][key] = (unsigned short)(vReg[i] >> 16);
    }
  };

  float m[8], l[8];
  v8f Oacc[4] = {};
#pragma unroll
  for (int r = 0; r < 8; ++r) { m[r] = NEG_BIG; l[r] = 0.0f; }

  const int nkb = (S + 63) >> 6;
  loadKV(0);
  storeKV(0);
  __syncthreads();

  for (int kb = 0; kb < nkb; ++kb) {
    int buf = kb & 1;
    if (kb + 1 < nkb) loadKV(kb + 1);          // loads fly during compute

    // ---- S = Q K^T : B-frag K-dim is d (contiguous in Ks rows)
    v8f sacc[4];
#pragma unroll
    for (int t = 0; t < 4; ++t) {
      const unsigned short* kp2 = &Ks[buf][t * 16 + lane16][0];
      v16bf b0 = cat8(*(const v8bf*)(kp2 + 16 * hi),
                      *(const v8bf*)(kp2 + 16 * hi + 8));
      v16bf b1 = cat8(*(const v8bf*)(kp2 + 32 + 16 * hi),
                      *(const v8bf*)(kp2 + 32 + 16 * hi + 8));
      v8f z = {};
      z = wmma_bf16(Qa[0], b0, z);
      z = wmma_bf16(Qa[1], b1, z);
      bool dead = (kb * 64 + t * 16 + lane16 >= S);   // invalid key column
#pragma unroll
      for (int r = 0; r < 8; ++r) sacc[t][r] = dead ? NEG_BIG : z[r] * scale;
    }

    // ---- online softmax; element r lives on row (r + 8*hi), 16-lane halves
#pragma unroll
    for (int r = 0; r < 8; ++r) {
      float vmax = fmaxf(fmaxf(sacc[0][r], sacc[1][r]),
                         fmaxf(sacc[2][r], sacc[3][r]));
#pragma unroll
      for (int msk = 8; msk >= 1; msk >>= 1)
        vmax = fmaxf(vmax, __shfl_xor(vmax, msk, 32));
      float mnew  = fmaxf(m[r], vmax);
      float alpha = __expf(m[r] - mnew);
      float rsum  = 0.0f;
#pragma unroll
      for (int t = 0; t < 4; ++t) {
        float p = __expf(sacc[t][r] - mnew);
        rsum += p;
        Ps[wave][r + 8 * hi][t * 16 + lane16] = f2bf(p);
      }
#pragma unroll
      for (int msk = 8; msk >= 1; msk >>= 1)
        rsum += __shfl_xor(rsum, msk, 32);
      l[r] = l[r] * alpha + rsum;
      m[r] = mnew;
#pragma unroll
      for (int t = 0; t < 4; ++t) Oacc[t][r] = Oacc[t][r] * alpha;
    }

    // ---- P fragments (A-layout) from per-wave staging: 2x ds_load_b128 each
    const unsigned short* pp = &Ps[wave][lane16][0];
    v16bf Pa0 = cat8(*(const v8bf*)(pp + 16 * hi),
                     *(const v8bf*)(pp + 8 + 16 * hi));
    v16bf Pa1 = cat8(*(const v8bf*)(pp + 32 + 16 * hi),
                     *(const v8bf*)(pp + 40 + 16 * hi));

    // ---- O += P @ V : B-frag K-dim is key (contiguous in transposed Vs rows)
#pragma unroll
    for (int t = 0; t < 4; ++t) {
      const unsigned short* vp = &Vs[buf][t * 16 + lane16][0];
      v16bf b0 = cat8(*(const v8bf*)(vp + 16 * hi),
                      *(const v8bf*)(vp + 16 * hi + 8));
      v16bf b1 = cat8(*(const v8bf*)(vp + 32 + 16 * hi),
                      *(const v8bf*)(vp + 32 + 16 * hi + 8));
      Oacc[t] = wmma_bf16(Pa0, b0, Oacc[t]);
      Oacc[t] = wmma_bf16(Pa1, b1, Oacc[t]);
    }

    if (kb + 1 < nkb) storeKV(buf ^ 1);
    __syncthreads();
  }

#pragma unroll
  for (int r = 0; r < 8; ++r) {
    int row = qbase + r + 8 * hi;
    if (row >= S) continue;
    float inv = (l[r] > 0.0f) ? (1.0f / l[r]) : 0.0f;
#pragma unroll
    for (int t = 0; t < 4; ++t)
      O[(size_t)row * DIM + hcol + t * 16 + lane16] = Oacc[t][r] * inv;
  }
}

// ---------------------------------------------------------------------------
// Small helper kernels
// ---------------------------------------------------------------------------
__global__ void prep_indices(const int* __restrict__ surf_idx, const int* __restrict__ velo_idx,
                             const int* __restrict__ closest, const int* __restrict__ labels,
                             const float* __restrict__ onion,
                             int* __restrict__ tok1_idx, int* __restrict__ velo_bucket)
{
  int i = blockIdx.x * 256 + threadIdx.x;
  if (i < S1) tok1_idx[i] = (i < FIRST_LAP) ? velo_idx[i] : surf_idx[i - FIRST_LAP];
  if (i < N_VELO) {
    int l = 0;
    for (int q = 1; q < ONION; ++q)
      if (onion[(size_t)q * N_VELO + i] > 0.5f) l = q;
    velo_bucket[i] = l * SC + labels[closest[i]];
  }
}

__global__ __launch_bounds__(256)
void pool_velo(const float* __restrict__ fx, const int* __restrict__ velo_idx,
               const int* __restrict__ bucket, float* __restrict__ tokens, int nchunks)
{
  __shared__ float acc[ONION * SC][256];
  int d = blockIdx.x * 256 + threadIdx.x;
  for (int b = 0; b < ONION * SC; ++b) acc[b][threadIdx.x] = 0.0f;
  int per = (N_VELO + nchunks - 1) / nchunks;
  int p0 = blockIdx.y * per;
  int p1 = p0 + per; if (p1 > N_VELO) p1 = N_VELO;
  for (int p = p0; p < p1; ++p)
    acc[bucket[p]][threadIdx.x] += fx[(size_t)velo_idx[p] * DIM + d];
  for (int b = 0; b < ONION * SC; ++b)
    atomicAdd(&tokens[(SC + b) * DIM + d], acc[b][threadIdx.x]);
}

__global__ __launch_bounds__(256)
void pool_surf(const float* __restrict__ fx, const int* __restrict__ surf_idx,
               const int* __restrict__ labels, float* __restrict__ tokens, int nchunks)
{
  __shared__ float acc[SC][256];
  int d = blockIdx.x * 256 + threadIdx.x;
  for (int b = 0; b < SC; ++b) acc[b][threadIdx.x] = 0.0f;
  int per = (SURF_KEEP + nchunks - 1) / nchunks;
  int p0 = blockIdx.y * per;
  int p1 = p0 + per; if (p1 > SURF_KEEP) p1 = SURF_KEEP;
  for (int j = p0; j < p1; ++j) {
    int sj = (j < 16) ? j : j + 96;       // concat(surf[:16], surf[112:])
    acc[labels[j]][threadIdx.x] += fx[(size_t)surf_idx[sj] * DIM + d];
  }
  for (int b = 0; b < SC; ++b)
    atomicAdd(&tokens[b * DIM + d], acc[b][threadIdx.x]);
}

__global__ void layernorm_tokens(const float* __restrict__ tokens,
                                 const float* __restrict__ gamma,
                                 const float* __restrict__ beta,
                                 float* __restrict__ out)
{
  int d = blockIdx.x * 256 + threadIdx.x;      // channel, 512 total
  float vals[NT];
  float mu = 0.0f;
#pragma unroll
  for (int t = 0; t < NT; ++t) {
    float s = (t < SC) ? (1.0f / SURF_KEEP) : (1.0f / N_VELO);
    vals[t] = tokens[t * DIM + d] * s;
    mu += vals[t];
  }
  mu *= (1.0f / NT);
  float var = 0.0f;
#pragma unroll
  for (int t = 0; t < NT; ++t) { float c = vals[t] - mu; var += c * c; }
  var *= (1.0f / NT);
  float rs = rsqrtf(var + 1e-5f);
#pragma unroll
  for (int t = 0; t < NT; ++t)
    out[t * DIM + d] = (vals[t] - mu) * rs * gamma[t] + beta[t];
}

__global__ void scatter_surf(const float* __restrict__ feat_surf, const float* __restrict__ fvc,
                             const float* __restrict__ O1, const int* __restrict__ surf_idx,
                             float* __restrict__ nf)
{
  int i = blockIdx.x;
  int row = surf_idx[i];
  for (int d = threadIdx.x; d < DIM; d += 256)
    nf[(size_t)row * DIM + d] = 0.5f * (feat_surf[(size_t)i * DIM + d] + fvc[(size_t)i * DIM + d])
                              + O1[(size_t)(FIRST_LAP + i) * DIM + d];
}

__global__ void scatter_velo(const float* __restrict__ feat_velo, const float* __restrict__ O1,
                             const int* __restrict__ velo_idx, float* __restrict__ nf)
{
  int i = blockIdx.x;
  int row = velo_idx[i];
  for (int d = threadIdx.x; d < DIM; d += 256) {
    float v = feat_velo[(size_t)i * DIM + d];
    if (i < FIRST_LAP) v += O1[(size_t)i * DIM + d];
    nf[(size_t)row * DIM + d] = v;
  }
}

// ---------------------------------------------------------------------------
extern "C" void kernel_launch(void* const* d_in, const int* in_sizes, int n_in,
                              void* d_out, int out_size, void* d_ws, size_t ws_size,
                              hipStream_t stream)
{
  const float* x     = (const float*)d_in[0];
  const float* onion = (const float*)d_in[1];
  const float* W_fx  = (const float*)d_in[2];
  const float* b_fx  = (const float*)d_in[3];
  const float* W_x   = (const float*)d_in[4];
  const float* b_x   = (const float*)d_in[5];
  const float* Wq1   = (const float*)d_in[6];
  const float* Wk1   = (const float*)d_in[7];
  const float* Wv1   = (const float*)d_in[8];
  const float* Wq    = (const float*)d_in[9];
  const float* Wk    = (const float*)d_in[10];
  const float* Wv    = (const float*)d_in[11];
  const float* gamma = (const float*)d_in[12];
  const float* beta  = (const float*)d_in[13];
  const float* W_ws  = (const float*)d_in[14];
  const float* b_ws  = (const float*)d_in[15];
  const float* W_vv  = (const float*)d_in[16];
  const float* b_vv  = (const float*)d_in[17];
  const float* W_vc  = (const float*)d_in[18];
  const float* b_vc  = (const float*)d_in[19];
  const float* W_out = (const float*)d_in[20];
  const float* b_out = (const float*)d_in[21];
  const int* surf_idx = (const int*)d_in[22];
  const int* velo_idx = (const int*)d_in[23];
  const int* closest  = (const int*)d_in[25];
  const int* labels   = (const int*)d_in[26];

  char* wsb = (char*)d_ws;
  size_t off = 0;
  auto alloc = [&](size_t bytes) -> void* {
    void* p = wsb + off;
    off = (off + bytes + 255) & ~(size_t)255;
    return p;
  };
  float* fx        = (float*)alloc((size_t)NTOT * DIM * 4);
  float* xm        = (float*)alloc((size_t)NTOT * DIM * 4);
  unsigned short* Q1 = (unsigned short*)alloc((size_t)S1 * DIM * 2);   // bf16
  unsigned short* K1 = (unsigned short*)alloc((size_t)S1 * DIM * 2);   // bf16
  unsigned short* V1 = (unsigned short*)alloc((size_t)S1 * DIM * 2);   // bf16
  float* O1        = (float*)alloc((size_t)S1 * DIM * 4);
  float* tokens    = (float*)alloc((size_t)NT * DIM * 4);
  float* tokn      = (float*)alloc((size_t)NT * DIM * 4);
  unsigned short* Qt = (unsigned short*)alloc((size_t)NT * DIM * 2);   // bf16
  unsigned short* Kt = (unsigned short*)alloc((size_t)NT * DIM * 2);   // bf16
  unsigned short* Vt = (unsigned short*)alloc((size_t)NT * DIM * 2);   // bf16
  float* Ot        = (float*)alloc((size_t)NT * DIM * 4);
  float* ws_       = (float*)alloc((size_t)N_SURF * 16 * 4);
  float* vv        = (float*)alloc((size_t)N_VELO * 64 * 4);
  float* vc        = (float*)alloc((size_t)N_SURF * 64 * 4);
  float* feat_surf = (float*)alloc((size_t)N_SURF * DIM * 4);
  float* fvc       = (float*)alloc((size_t)N_SURF * DIM * 4);
  float* feat_velo = (float*)alloc((size_t)N_VELO * DIM * 4);
  float* nf        = (float*)alloc((size_t)NTOT * DIM * 4);
  int* tok1_idx    = (int*)alloc((size_t)S1 * 4);
  int* velo_bucket = (int*)alloc((size_t)N_VELO * 4);

  auto gemm = [&](const float* A, const int* ridx, int lda, const float* B, int ldb,
                  const float* bias, void* C, int ldc, int M, int N, int K, int obf) {
    dim3 g((N + BN - 1) / BN, (M + BM - 1) / BM);
    gemm_wmma<<<g, 256, 0, stream>>>(A, ridx, lda, B, ldb, bias, C, ldc, M, N, K, obf);
  };

  // 0) index prep
  prep_indices<<<(N_VELO + 255) / 256, 256, 0, stream>>>(
      surf_idx, velo_idx, closest, labels, onion, tok1_idx, velo_bucket);

  // 1) fx = x @ W_fx + b_fx ; xm = x @ W_x + b_x
  gemm(x, nullptr, DIM, W_fx, DIM, b_fx, fx, DIM, NTOT, DIM, DIM, 0);
  gemm(x, nullptr, DIM, W_x,  DIM, b_x,  xm, DIM, NTOT, DIM, DIM, 0);

  // 2) per-head QKV projections for attention-1 (gathered rows, bf16 output)
  for (int h = 0; h < HEADS; ++h) {
    gemm(fx + h * DH, tok1_idx, DIM, Wq1, DH, nullptr, Q1 + h * DH, DIM, S1, DH, DH, 1);
    gemm(fx + h * DH, tok1_idx, DIM, Wk1, DH, nullptr, K1 + h * DH, DIM, S1, DH, DH, 1);
    gemm(fx + h * DH, tok1_idx, DIM, Wv1, DH, nullptr, V1 + h * DH, DIM, S1, DH, DH, 1);
  }

  // 3) flash attention over 4182 tokens, 8 heads
  flash_attn<<<dim3((S1 + 63) / 64, HEADS), 128, 0, stream>>>(Q1, K1, V1, O1, S1, 0.125f);

  // 4) token pooling
  hipMemsetAsync(tokens, 0, (size_t)NT * DIM * 4, stream);
  pool_velo<<<dim3(2, 28), 256, 0, stream>>>(fx, velo_idx, velo_bucket, tokens, 28);
  pool_surf<<<dim3(2, 4), 256, 0, stream>>>(fx, surf_idx, labels, tokens, 4);

  // 5) layernorm over the token axis (pool scaling folded in)
  layernorm_tokens<<<2, 256, 0, stream>>>(tokens, gamma, beta, tokn);

  // 6) attention-2 over 64 pooled tokens
  for (int h = 0; h < HEADS; ++h) {
    gemm(tokn + h * DH, nullptr, DIM, Wq, DH, nullptr, Qt + h * DH, DIM, NT, DH, DH, 1);
    gemm(tokn + h * DH, nullptr, DIM, Wk, DH, nullptr, Kt + h * DH, DIM, NT, DH, DH, 1);
    gemm(tokn + h * DH, nullptr, DIM, Wv, DH, nullptr, Vt + h * DH, DIM, NT, DH, DH, 1);
  }
  flash_attn<<<dim3(1, HEADS), 128, 0, stream>>>(Qt, Kt, Vt, Ot, NT, 0.125f);

  // 7) low-rank feature products
  gemm(xm, surf_idx, DIM, W_ws, SC, b_ws, ws_, 16, N_SURF, SC, DIM, 0);
  gemm(ws_, nullptr, 16, Ot, DIM, nullptr, feat_surf, DIM, N_SURF, DIM, SC, 0);
  gemm(xm, velo_idx, DIM, W_vv, ONION * SC, b_vv, vv, 64, N_VELO, ONION * SC, DIM, 0);
  gemm(vv, nullptr, 64, Ot + SC * DIM, DIM, nullptr, feat_velo, DIM, N_VELO, DIM, ONION * SC, 0);
  gemm(xm, surf_idx, DIM, W_vc, ONION * SC, b_vc, vc, 64, N_SURF, ONION * SC, DIM, 0);
  gemm(vc, nullptr, 64, Ot + SC * DIM, DIM, nullptr, fvc, DIM, N_SURF, DIM, ONION * SC, 0);

  // 8) reassemble new_feature
  scatter_surf<<<N_SURF, 256, 0, stream>>>(feat_surf, fvc, O1, surf_idx, nf);
  scatter_velo<<<N_VELO, 256, 0, stream>>>(feat_velo, O1, velo_idx, nf);

  // 9) output projection
  gemm(nf, nullptr, DIM, W_out, DIM, b_out, d_out, DIM, NTOT, DIM, DIM, 0);

  (void)in_sizes; (void)n_in; (void)out_size; (void)ws_size;
}
